// SpatialTransformerPredictor_91027536871561
// MI455X (gfx1250) — matrix operand order repaired and checked by reading, not compiled, not run on hardware
//
#include <hip/hip_runtime.h>
#include <hip/hip_bf16.h>

typedef __attribute__((ext_vector_type(16))) _Float16 v16h;
typedef __attribute__((ext_vector_type(8)))  float    v8f;
typedef __attribute__((ext_vector_type(4)))  float    v4f;

constexpr int B_ = 16, N_ = 10, H_ = 400, W_ = 400;
constexpr int HW = H_ * W_;            // 160000
constexpr int CHUNKS   = 25;           // chunks per batch image
constexpr int CHUNK    = HW / CHUNKS;  // 6400
constexpr int WAVES    = 8;            // 256-thread block = 8 wave32
constexpr int PER_WAVE = CHUNK / WAVES;   // 800
constexpr int ITERS    = PER_WAVE / 32;   // 25 WMMA steps / wave
static_assert(CHUNKS * CHUNK == HW, "exact tiling");
static_assert(WAVES * PER_WAVE == CHUNK, "exact tiling");
static_assert(ITERS * 32 == PER_WAVE, "exact tiling");

// ---------------------------------------------------------------------------
// Kernel 0: zero output + workspace accumulators (harness poisons them).
// ---------------------------------------------------------------------------
__global__ void init_kernel(float* __restrict__ out, float* __restrict__ ws) {
    int i = blockIdx.x * blockDim.x + threadIdx.x;
    if (i < B_ * N_ * 2) out[i] = 0.0f;
    if (i < B_ + B_ * N_) ws[i] = 0.0f;
}

// ---------------------------------------------------------------------------
// Kernel 1: overlap[b,n] = sum_hw om[b,n,hw] * tm[b,hw]  via v_wmma_f32_16x16x32_f16
//   A (16x32)  = tm chunk replicated across the 16 M-rows
//   B (32x16)  = om rows as columns (n = lane%16; n>=10 zeroed)
//   D[m,n]     = dot(tm_chunk, om[n]_chunk)  (identical over m)
// Exact: masks are {0,1}; f16 conversion exact; f32 accum of integers < 2^24.
// ---------------------------------------------------------------------------
__global__ __launch_bounds__(256) void overlap_wmma_kernel(
    const float* __restrict__ om, const float* __restrict__ tm,
    float* __restrict__ out)
{
    const int tid   = threadIdx.x;
    const int lane  = tid & 31;
    const int wave  = tid >> 5;
    const int b     = blockIdx.x / CHUNKS;
    const int chunk = blockIdx.x % CHUNKS;

    const int  ncol = lane & 15;          // B/D column index (object n)
    const int  hi   = lane >> 4;          // half-wave select
    const int  kb_b = hi * 16;            // B-fragment K base (contiguous k = 16*hi + s)
    const int  kb_a = hi * 16;            // A-fragment K base (matches B)

    const int   nrow   = (ncol < N_) ? ncol : 0;
    const float nvalid = (ncol < N_) ? 1.0f : 0.0f;   // zero padded columns

    const float* __restrict__ tmB = tm + (size_t)b * HW;
    const float* __restrict__ omR = om + ((size_t)(b * N_ + nrow)) * HW;

    v8f c = {};
    int k0 = chunk * CHUNK + wave * PER_WAVE;   // always a multiple of 32

    for (int i = 0; i < ITERS; ++i, k0 += 32) {
        // ---- B fragment: 16 consecutive om floats (read-once -> non-temporal)
        const v4f* pb = (const v4f*)(omR + k0 + kb_b);
        v4f q0 = __builtin_nontemporal_load(pb + 0);
        v4f q1 = __builtin_nontemporal_load(pb + 1);
        v4f q2 = __builtin_nontemporal_load(pb + 2);
        v4f q3 = __builtin_nontemporal_load(pb + 3);
        v16h bf;
        bf[0]  = (_Float16)(q0[0] * nvalid); bf[1]  = (_Float16)(q0[1] * nvalid);
        bf[2]  = (_Float16)(q0[2] * nvalid); bf[3]  = (_Float16)(q0[3] * nvalid);
        bf[4]  = (_Float16)(q1[0] * nvalid); bf[5]  = (_Float16)(q1[1] * nvalid);
        bf[6]  = (_Float16)(q1[2] * nvalid); bf[7]  = (_Float16)(q1[3] * nvalid);
        bf[8]  = (_Float16)(q2[0] * nvalid); bf[9]  = (_Float16)(q2[1] * nvalid);
        bf[10] = (_Float16)(q2[2] * nvalid); bf[11] = (_Float16)(q2[3] * nvalid);
        bf[12] = (_Float16)(q3[0] * nvalid); bf[13] = (_Float16)(q3[1] * nvalid);
        bf[14] = (_Float16)(q3[2] * nvalid); bf[15] = (_Float16)(q3[3] * nvalid);

        // ---- A fragment: tm, replicated across M rows; per-lane K pattern:
        //      lane<16: K {0..15},  lane>=16: K {16..31}
        const float* pa = tmB + k0 + kb_a;
        v4f t0 = *(const v4f*)(pa + 0);
        v4f t1 = *(const v4f*)(pa + 4);
        v4f t2 = *(const v4f*)(pa + 8);
        v4f t3 = *(const v4f*)(pa + 12);
        v16h af;
        af[0]  = (_Float16)t0[0]; af[1]  = (_Float16)t0[1];
        af[2]  = (_Float16)t0[2]; af[3]  = (_Float16)t0[3];
        af[4]  = (_Float16)t1[0]; af[5]  = (_Float16)t1[1];
        af[6]  = (_Float16)t1[2]; af[7]  = (_Float16)t1[3];
        af[8]  = (_Float16)t2[0]; af[9]  = (_Float16)t2[1];
        af[10] = (_Float16)t2[2]; af[11] = (_Float16)t2[3];
        af[12] = (_Float16)t3[0]; af[13] = (_Float16)t3[1];
        af[14] = (_Float16)t3[2]; af[15] = (_Float16)t3[3];

        c = __builtin_amdgcn_wmma_f32_16x16x32_f16(
                /*neg_a=*/false, af, /*neg_b=*/false, bf,
                /*c_mod=*/(short)0, c, /*reuse_a=*/false, /*reuse_b=*/false);
    }

    // c[0] holds D[M=0][N=lane] for lanes 0..15 -> partial overlap for n=lane.
    if (lane < N_)
        atomicAdd(&out[(b * N_ + lane) * 2 + 0], c[0]);
}

// ---------------------------------------------------------------------------
// Kernel 2: tgt_area[b] = sum_hw tm[b,hw]   (block reduce + one atomic/block)
// ---------------------------------------------------------------------------
__global__ __launch_bounds__(256) void tgt_area_kernel(
    const float* __restrict__ tm, float* __restrict__ ws_tgt)
{
    __shared__ float red[256];
    const int b     = blockIdx.x / CHUNKS;
    const int chunk = blockIdx.x % CHUNKS;
    const float* __restrict__ p = tm + (size_t)b * HW + chunk * CHUNK;
    float s = 0.0f;
    for (int i = threadIdx.x; i < CHUNK; i += 256) s += p[i];
    red[threadIdx.x] = s;
    __syncthreads();
    for (int off = 128; off > 0; off >>= 1) {
        if (threadIdx.x < off) red[threadIdx.x] += red[threadIdx.x + off];
        __syncthreads();
    }
    if (threadIdx.x == 0) atomicAdd(&ws_tgt[b], red[0]);
}

// ---------------------------------------------------------------------------
// Kernel 3: inter[b,n] = sum of tm over the clamped bbox rectangle.
// One block per (b,n); tm is L2-resident (10 MB << 192 MB).
// ---------------------------------------------------------------------------
__global__ __launch_bounds__(256) void inter_kernel(
    const float* __restrict__ bboxes, const float* __restrict__ tm,
    float* __restrict__ ws_inter)
{
    __shared__ float red[256];
    const int b = blockIdx.x / N_;
    const int n = blockIdx.x % N_;
    const float* bb = bboxes + (size_t)(b * N_ + n) * 4;
    int x1 = (int)floorf(bb[0]);
    int y1 = (int)floorf(bb[1]);
    int x2 = (int)floorf(bb[2]);
    int y2 = (int)floorf(bb[3]);
    x1 = min(max(x1, 0), W_); x2 = min(max(x2, 0), W_);
    y1 = min(max(y1, 0), H_); y2 = min(max(y2, 0), H_);
    const int rw = max(0, x2 - x1);
    const int rh = max(0, y2 - y1);
    const int total = rw * rh;

    float s = 0.0f;
    const float* __restrict__ tmB = tm + (size_t)b * HW;
    for (int i = threadIdx.x; i < total; i += 256) {
        int y = y1 + i / rw;
        int x = x1 + i % rw;
        s += tmB[y * W_ + x];
    }
    red[threadIdx.x] = s;
    __syncthreads();
    for (int off = 128; off > 0; off >>= 1) {
        if (threadIdx.x < off) red[threadIdx.x] += red[threadIdx.x + off];
        __syncthreads();
    }
    if (threadIdx.x == 0) ws_inter[b * N_ + n] = red[0];
}

// ---------------------------------------------------------------------------
// Kernel 4: iou = inter / (box_area + tgt_area - inter + 1e-8); write out[...,1]
// ---------------------------------------------------------------------------
__global__ void finalize_kernel(
    const float* __restrict__ bboxes, const float* __restrict__ ws_tgt,
    const float* __restrict__ ws_inter, float* __restrict__ out)
{
    int idx = blockIdx.x * blockDim.x + threadIdx.x;
    if (idx >= B_ * N_) return;
    const int b = idx / N_;
    const float* bb = bboxes + (size_t)idx * 4;
    int x1 = (int)floorf(bb[0]);
    int y1 = (int)floorf(bb[1]);
    int x2 = (int)floorf(bb[2]);
    int y2 = (int)floorf(bb[3]);
    x1 = min(max(x1, 0), W_); x2 = min(max(x2, 0), W_);
    y1 = min(max(y1, 0), H_); y2 = min(max(y2, 0), H_);
    const float cols = (float)max(0, x2 - x1);
    const float rows = (float)max(0, y2 - y1);
    const float box_area = rows * cols;
    const float inter = ws_inter[idx];
    const float tgt   = ws_tgt[b];
    out[idx * 2 + 1] = inter / (box_area + tgt - inter + 1e-8f);
}

// ---------------------------------------------------------------------------
extern "C" void kernel_launch(void* const* d_in, const int* in_sizes, int n_in,
                              void* d_out, int out_size, void* d_ws, size_t ws_size,
                              hipStream_t stream) {
    const float* bboxes = (const float*)d_in[0];   // (B,N,4)
    const float* om     = (const float*)d_in[1];   // (B,N,1,H,W)
    const float* tm     = (const float*)d_in[2];   // (B,1,H,W)
    float* out = (float*)d_out;                    // (B,N,2)

    float* ws_tgt   = (float*)d_ws;        // [B_]
    float* ws_inter = ws_tgt + B_;         // [B_*N_]

    init_kernel<<<1, 512, 0, stream>>>(out, ws_tgt);
    overlap_wmma_kernel<<<B_ * CHUNKS, 256, 0, stream>>>(om, tm, out);
    tgt_area_kernel<<<B_ * CHUNKS, 256, 0, stream>>>(tm, ws_tgt);
    inter_kernel<<<B_ * N_, 256, 0, stream>>>(bboxes, tm, ws_inter);
    finalize_kernel<<<1, 256, 0, stream>>>(bboxes, ws_tgt, ws_inter, out);
}
